// NOC_83210696393089
// MI455X (gfx1250) — matrix operand order repaired and checked by reading, not compiled, not run on hardware
//
#include <hip/hip_runtime.h>
#include <hip/hip_bf16.h>

// ---------------------------------------------------------------------------
// CDNA5 (gfx1250) fused persistent implementation of the sequential
// clustering scan. All per-step GEMMs -> v_wmma_f32_16x16x32_bf16 (f32 acc).
// 8 blocks x 16 chains (M=16 == WMMA tile), 512 threads = 16 wave32.
// Phase E stages d_part through LDS in 32-row slabs (16x L2-traffic cut),
// using GLOBAL_LOAD_ASYNC_TO_LDS_B128 (builtin confirmed present).
// ---------------------------------------------------------------------------

typedef __bf16 bf16_t;
typedef __attribute__((ext_vector_type(16))) __bf16 v16bf;
typedef __attribute__((ext_vector_type(8)))  __bf16 v8bf;
typedef __attribute__((ext_vector_type(8)))  float  v8f;

#define N_PTS      1024
#define S_CH       128
#define K_STEPS    32
#define CH_BLK     16
#define XCOLS      1152   // [mu 0:128 | A 128:384 | Uk 384:640 | G 640:1152]
#define DSTRIDE    260    // padded f32 stride for the d_part slab (bank spread)

// ---- LDS layout (bytes) ----
#define OFF_X      0        // 16x1152 bf16  = 36864
#define OFF_SPART  36864    // 16x256  f32   = 16384
#define OFF_G      53248    // 16x512  f32   = 32768
#define OFF_NHS    86016    // 16x256  bf16  =  8192
#define OFF_MBITS  94208    // 16x32   u32   =  2048
#define OFF_NBITS  96256    // 16x32   u32   =  2048
#define OFF_ACT    98304    // 16 f32
#define OFF_IMS    98368    // 16 f32
#define OFF_INC    98432    // 16 f32
#define OFF_W2     98560    // 256 f32 = 1024
#define OFF_WOR    99584    // 32 u32  = 128
#define OFF_DBUF   99712    // 32x260 f32 = 33280
#define SMEM_BYTES 132992

// ---- async global->LDS path (guarded; clean fallback keeps us compiling) --
#if defined(__has_builtin)
#  if __has_builtin(__builtin_amdgcn_global_load_async_to_lds_b128)
#    define HAVE_ASYNC_LDS 1
#  endif
#endif
#ifndef HAVE_ASYNC_LDS
#  define HAVE_ASYNC_LDS 0
#endif

typedef int v4i_t __attribute__((ext_vector_type(4)));

#if HAVE_ASYNC_LDS
typedef v4i_t __attribute__((address_space(1))) g_v4i;   // global int4*
typedef v4i_t __attribute__((address_space(3))) l_v4i;   // LDS int4*
#  if __has_builtin(__builtin_amdgcn_s_wait_asynccnt)
#    define ASYNC_WAIT() __builtin_amdgcn_s_wait_asynccnt(0)
#  else
#    define ASYNC_WAIT() asm volatile("s_wait_asynccnt 0x0" ::: "memory")
#  endif
#endif

// copy 16 bytes global -> LDS (async when available)
static __device__ __forceinline__ void copy16_g2l(const float* gp, float* lp) {
#if HAVE_ASYNC_LDS
  __builtin_amdgcn_global_load_async_to_lds_b128(
      (g_v4i*)(unsigned long long)(const void*)gp,
      (l_v4i*)(unsigned)(unsigned long long)(void*)lp, 0, 0);
#else
  *(float4*)lp = *(const float4*)gp;
#endif
}

static __device__ __forceinline__ v8f vzero8() {
  v8f z;
#pragma unroll
  for (int i = 0; i < 8; ++i) z[i] = 0.f;
  return z;
}

static __device__ __forceinline__ v8f wmma_bf16(v16bf a, v16bf b, v8f c) {
  return __builtin_amdgcn_wmma_f32_16x16x32_bf16(false, a, false, b,
                                                 (short)0, c, false, false);
}

// A operand (16x32 bf16 tile), source row-major with K contiguous.
static __device__ __forceinline__ v16bf load_a16(const bf16_t* base, int stride, int lane) {
  const bf16_t* p = base + (lane & 15) * stride + ((lane < 16) ? 0 : 8);
  v8bf lo = *(const v8bf*)p;
  v8bf hi = *(const v8bf*)(p + 16);
  v16bf r;
#pragma unroll
  for (int i = 0; i < 8; ++i) { r[i] = lo[i]; r[i + 8] = hi[i]; }
  return r;
}

// B operand (32x16 bf16 tile) from a transposed (NxK, K contiguous) matrix.
static __device__ __forceinline__ v16bf load_b16(const bf16_t* base, int stride, int lane) {
  const bf16_t* p = base + (lane & 15) * stride + ((lane < 16) ? 0 : 8);
  v8bf lo = *(const v8bf*)p;
  v8bf hi = *(const v8bf*)(p + 16);
  v16bf r;
#pragma unroll
  for (int i = 0; i < 8; ++i) { r[i] = lo[i]; r[i + 8] = hi[i]; }
  return r;
}

// A operand generated from a 32-bit mask word (one word == one 32-K slab).
static __device__ __forceinline__ v16bf mask_a16(unsigned word, bf16_t inv, int lane) {
  const int sh = (lane < 16) ? 0 : 8;
  v16bf r;
#pragma unroll
  for (int e = 0; e < 16; ++e) {
    const int ko = ((e < 8) ? e : (e + 8)) + sh;
    r[e] = ((word >> ko) & 1u) ? inv : (bf16_t)0.f;
  }
  return r;
}

// ---------------------------------------------------------------------------
// Prep: bf16 casts + weight transposes + cs init (one-time, L2-resident).
// ---------------------------------------------------------------------------
__global__ void prep_kernel(const float* enc, const float* Wh, const float* Wu,
                            const float* Wpz, const float* Wphi1, const float* Wg,
                            bf16_t* encB, bf16_t* WhT, bf16_t* WuT, bf16_t* W1dT,
                            bf16_t* WpzTr, bf16_t* W1restT, bf16_t* WgT,
                            int* cs_out) {
  const int tid = blockIdx.x * blockDim.x + threadIdx.x;
  const int nt  = gridDim.x * blockDim.x;
  for (int i = tid; i < 1024 * 256; i += nt) encB[i] = (bf16_t)enc[i];
  for (int i = tid; i < 256 * 256; i += nt) {
    const int n = i >> 8, k = i & 255;
    WhT[i]  = (bf16_t)Wh[k * 256 + n];
    WuT[i]  = (bf16_t)Wu[k * 256 + n];
    W1dT[i] = (bf16_t)Wphi1[k * 256 + n];          // rows 0..255 of Wphi1
  }
  for (int i = tid; i < 128 * 1024; i += nt) {      // WpzTr[z][j] = Wpz[j][z]
    const int z = i >> 10, j = i & 1023;
    WpzTr[i] = (bf16_t)Wpz[j * 256 + z];
  }
  for (int i = tid; i < 256 * 1152; i += nt) {      // W1restT[h][j] = Wphi1[256+j][h]
    const int h = i / 1152, j = i - h * 1152;
    W1restT[i] = (bf16_t)Wphi1[(256 + j) * 256 + h];
  }
  for (int i = tid; i < 512 * 256; i += nt) {       // WgT[g][h] = Wg[h][g]
    const int g = i >> 8, h = i & 255;
    WgT[i] = (bf16_t)Wg[h * 512 + g];
  }
  for (int i = tid; i < 128 * 1024; i += nt) cs_out[i] = -1;
}

// ---------------------------------------------------------------------------
// Precompute GEMM: C(1024x256) = encB(1024x256) @ W (via WT, 256x256).
// mode 0: store f32 row-major (d_part). mode 1: store bf16 transposed (NxM).
// ---------------------------------------------------------------------------
__global__ void __launch_bounds__(256)
gemm_ew(const bf16_t* __restrict__ A, const bf16_t* __restrict__ BT,
        float* outF, bf16_t* outT, int mode) {
  const int lane = threadIdx.x & 31, wave = threadIdx.x >> 5;
  const int tile = blockIdx.x * 8 + wave;   // 1024 tiles
  const int mt = tile >> 4, nt = tile & 15; // 64 x 16
  v8f acc = vzero8();
#pragma unroll
  for (int kt = 0; kt < 8; ++kt) {
    v16bf a = load_a16(A  + (mt * 16) * 256 + kt * 32, 256, lane);
    v16bf b = load_b16(BT + (nt * 16) * 256 + kt * 32, 256, lane);
    acc = wmma_bf16(a, b, acc);
  }
  const int n  = nt * 16 + (lane & 15);
  const int mb = (lane < 16) ? 0 : 8;
#pragma unroll
  for (int r = 0; r < 8; ++r) {
    const int m = mt * 16 + mb + r;
    if (mode == 0) outF[m * 256 + n] = acc[r];
    else           outT[n * 1024 + m] = (bf16_t)acc[r];
  }
}

// ---------------------------------------------------------------------------
// Main persistent kernel: 8 blocks x 16 chains, 32 sequential steps.
// ---------------------------------------------------------------------------
__global__ void __launch_bounds__(512)
cluster_kernel(const float* __restrict__ enc, const float* __restrict__ Wphi2,
               const float* __restrict__ gumbel, const float* __restrict__ unif,
               const bf16_t* __restrict__ usT, const bf16_t* __restrict__ hsT,
               const float* __restrict__ d_part,
               const bf16_t* __restrict__ WpzTr, const bf16_t* __restrict__ W1restT,
               const bf16_t* __restrict__ WgT,
               int* __restrict__ cs_out, float* __restrict__ G_out) {
  extern __shared__ char smem[];
  bf16_t*   X       = (bf16_t*)(smem + OFF_X);
  float*    spart   = (float*)(smem + OFF_SPART);
  float*    Gl      = (float*)(smem + OFF_G);
  bf16_t*   nhs     = (bf16_t*)(smem + OFF_NHS);
  unsigned* mbits   = (unsigned*)(smem + OFF_MBITS);
  unsigned* nbits   = (unsigned*)(smem + OFF_NBITS);
  float*    activeF = (float*)(smem + OFF_ACT);
  float*    invMs   = (float*)(smem + OFF_IMS);
  float*    invNc   = (float*)(smem + OFF_INC);
  float*    w2      = (float*)(smem + OFF_W2);
  unsigned* wordOr  = (unsigned*)(smem + OFF_WOR);
  float*    dbuf    = (float*)(smem + OFF_DBUF);

  const int tid = threadIdx.x;
  const int lane = tid & 31, wave = tid >> 5;
  const int sBase = blockIdx.x * CH_BLK;

  for (int i = tid; i < 16 * 512; i += 512) Gl[i] = 0.f;
  for (int i = tid; i < 16 * 32; i += 512) mbits[i] = 0xFFFFFFFFu;
  for (int i = tid; i < 256; i += 512) w2[i] = Wphi2[i];
  __syncthreads();

  for (int k = 0; k < K_STEPS; ++k) {
    // ---- Phase A: per-chain scalar work (wave w <-> chain w) --------------
    {
      const int c = wave;
      const int s = sBase + c;
      int pc = __popc(mbits[c * 32 + lane]);
      for (int off = 16; off; off >>= 1) pc += __shfl_xor(pc, off, 32);
      const int activ = (pc > 0) ? 1 : 0;

      // masked gumbel argmax (lane scans n = i*32 + lane)
      float best = -3.4e38f; int bidx = 0;
      const float* grow = gumbel + (k * S_CH + s) * N_PTS;
      for (int i = 0; i < 32; ++i) {
        const unsigned w = mbits[c * 32 + i];
        if ((w >> lane) & 1u) {
          const int n = i * 32 + lane;
          const float v = grow[n];
          if (v > best || (v == best && n < bidx)) { best = v; bidx = n; }
        }
      }
      for (int off = 16; off; off >>= 1) {
        const float ob = __shfl_xor(best, off, 32);
        const int   oi = __shfl_xor(bidx, off, 32);
        if (ob > best || (ob == best && oi < bidx)) { best = ob; bidx = oi; }
      }
      const int anch = activ ? bidx : 0;   // argmax of all -inf -> 0

      nbits[c * 32 + lane] = 0u;           // aoh seed (wave-synchronous)
      if (lane == 0) {
        if (activ) {
          mbits[c * 32 + (anch >> 5)] &= ~(1u << (anch & 31));
          nbits[c * 32 + (anch >> 5)]  =  (1u << (anch & 31));
          cs_out[s * N_PTS + anch] = k;
        }
        activeF[c] = activ ? 1.f : 0.f;
        int ms = pc - activ; if (ms < 1) ms = 1;
        invMs[c] = 1.f / (float)ms;
      }
      // prefetch noise streams we will need soon (global_prefetch_b8)
      __builtin_prefetch(unif + (k * S_CH + s) * N_PTS + lane * 32, 0, 1);
      if (k + 1 < K_STEPS)
        __builtin_prefetch(gumbel + ((k + 1) * S_CH + s) * N_PTS + lane * 32, 0, 1);

      // fill X: A row (anchor enc row) and current G, bf16
      const float* arow = enc + (size_t)anch * 256;
      for (int j = lane; j < 256; j += 32) X[c * XCOLS + 128 + j] = (bf16_t)arow[j];
      for (int j = lane; j < 512; j += 32) X[c * XCOLS + 640 + j] = (bf16_t)Gl[c * 512 + j];
    }
    __syncthreads();

    // ---- Phase B: Uk = mask/msum @ us  (M=16,N=256,K=1024) ----------------
    {
      const int nt = wave;
      const int r  = lane & 15;
      const bf16_t invb = (bf16_t)invMs[r];
      v8f acc = vzero8();
      for (int kt = 0; kt < 32; ++kt) {
        v16bf a = mask_a16(mbits[r * 32 + kt], invb, lane);
        v16bf b = load_b16(usT + (size_t)(nt * 16) * 1024 + kt * 32, 1024, lane);
        acc = wmma_bf16(a, b, acc);
      }
      const int n  = nt * 16 + (lane & 15);
      const int mb = (lane < 16) ? 0 : 8;
#pragma unroll
      for (int rr = 0; rr < 8; ++rr) X[(rr + mb) * XCOLS + 384 + n] = (bf16_t)acc[rr];
    }
    __syncthreads();

    // ---- Phase C: mu = [A|Uk|G] @ Wpz[:, :128]  (M=16,N=128,K=1024) -------
    if (wave < 8) {
      const int nt = wave;
      v8f acc = vzero8();
      for (int kt = 0; kt < 32; ++kt) {
        v16bf a = load_a16(X + 128 + kt * 32, XCOLS, lane);
        v16bf b = load_b16(WpzTr + (size_t)(nt * 16) * 1024 + kt * 32, 1024, lane);
        acc = wmma_bf16(a, b, acc);
      }
      const int n  = nt * 16 + (lane & 15);
      const int mb = (lane < 16) ? 0 : 8;
#pragma unroll
      for (int rr = 0; rr < 8; ++rr) X[(rr + mb) * XCOLS + n] = (bf16_t)acc[rr];
    }
    __syncthreads();

    // ---- Phase D: s_part = [mu|A|Uk|G] @ W1_rest  (M=16,N=256,K=1152) -----
    {
      const int nt = wave;
      v8f acc = vzero8();
      for (int kt = 0; kt < 36; ++kt) {
        v16bf a = load_a16(X + kt * 32, XCOLS, lane);
        v16bf b = load_b16(W1restT + (size_t)(nt * 16) * 1152 + kt * 32, 1152, lane);
        acc = wmma_bf16(a, b, acc);
      }
      const int n  = nt * 16 + (lane & 15);
      const int mb = (lane < 16) ? 0 : 8;
#pragma unroll
      for (int rr = 0; rr < 8; ++rr) spart[(rr + mb) * 256 + n] = acc[rr];
    }
    __syncthreads();

    // ---- Phase E: relu-gated logits + Bernoulli sampling ------------------
    // Stage d_part in 32-row LDS slabs: each row fetched once from L2 and
    // reused by all 16 chains; empty slabs (mask OR == 0) skipped entirely.
    if (tid < 32) {
      unsigned o = 0u;
#pragma unroll
      for (int c2 = 0; c2 < 16; ++c2) o |= mbits[c2 * 32 + tid];
      wordOr[tid] = o;
    }
    __syncthreads();
    {
      const int c = wave;
      const int s = sBase + c;
      const float* srow = spart + c * 256;
      const float actc = activeF[c];
      const float* urow = unif + (k * S_CH + s) * N_PTS;
      for (int slab = 0; slab < 32; ++slab) {
        if (wordOr[slab] == 0u) continue;          // block-uniform skip
        for (int q = tid; q < 32 * 64; q += 512) { // 32 rows x 64 float4
          const int row = q >> 6, c4 = (q & 63) << 2;
          copy16_g2l(d_part + (size_t)(slab * 32 + row) * 256 + c4,
                     dbuf + row * DSTRIDE + c4);
        }
#if HAVE_ASYNC_LDS
        ASYNC_WAIT();
#endif
        __syncthreads();
        const unsigned mword = mbits[c * 32 + slab];
        if (((mword >> lane) & 1u) && actc != 0.f) {
          const int n = slab * 32 + lane;
          const float* dr = dbuf + lane * DSTRIDE;
          float acc = 0.f;
#pragma unroll 4
          for (int h = 0; h < 256; h += 4) {
            const float4 d  = *(const float4*)(dr + h);
            const float4 sp = *(const float4*)(srow + h);
            const float4 ww = *(const float4*)(w2 + h);
            acc += ww.x * fmaxf(d.x + sp.x, 0.f);
            acc += ww.y * fmaxf(d.y + sp.y, 0.f);
            acc += ww.z * fmaxf(d.z + sp.z, 0.f);
            acc += ww.w * fmaxf(d.w + sp.w, 0.f);
          }
          const float prob = 1.f / (1.f + __expf(-acc));
          if (urow[n] < prob) {
            atomicOr (&nbits[c * 32 + slab],  (1u << lane));
            atomicAnd(&mbits[c * 32 + slab], ~(1u << lane));
            cs_out[s * N_PTS + n] = k;
          }
        }
        __syncthreads();   // dbuf reused next slab
      }
    }
    __syncthreads();

    // ---- Phase E2: 1/max(sum(new_cluster),1) ------------------------------
    {
      const int c = wave;
      int pc = __popc(nbits[c * 32 + lane]);
      for (int off = 16; off; off >>= 1) pc += __shfl_xor(pc, off, 32);
      if (lane == 0) invNc[c] = 1.f / (float)((pc < 1) ? 1 : pc);
    }
    __syncthreads();

    // ---- Phase F: newHs = newC/ncsum @ hs  (M=16,N=256,K=1024) ------------
    {
      const int nt = wave;
      const int r  = lane & 15;
      const bf16_t invb = (bf16_t)invNc[r];
      v8f acc = vzero8();
      for (int kt = 0; kt < 32; ++kt) {
        v16bf a = mask_a16(nbits[r * 32 + kt], invb, lane);
        v16bf b = load_b16(hsT + (size_t)(nt * 16) * 1024 + kt * 32, 1024, lane);
        acc = wmma_bf16(a, b, acc);
      }
      const int n  = nt * 16 + (lane & 15);
      const int mb = (lane < 16) ? 0 : 8;
#pragma unroll
      for (int rr = 0; rr < 8; ++rr) nhs[(rr + mb) * 256 + n] = (bf16_t)acc[rr];
    }
    __syncthreads();

    // ---- Phase G: G += (newHs @ Wg) * active  (M=16,N=512,K=256) ----------
    for (int t = 0; t < 2; ++t) {
      const int nt = wave + t * 16;
      v8f acc = vzero8();
#pragma unroll
      for (int kt = 0; kt < 8; ++kt) {
        v16bf a = load_a16(nhs + kt * 32, 256, lane);
        v16bf b = load_b16(WgT + (size_t)(nt * 16) * 256 + kt * 32, 256, lane);
        acc = wmma_bf16(a, b, acc);
      }
      const int n  = nt * 16 + (lane & 15);
      const int mb = (lane < 16) ? 0 : 8;
#pragma unroll
      for (int rr = 0; rr < 8; ++rr) {
        const int m = rr + mb;
        Gl[m * 512 + n] += acc[rr] * activeF[m];
      }
    }
    __syncthreads();
  }

  // ---- Emit G ----
  for (int i = tid; i < 16 * 512; i += 512) {
    const int c = i >> 9, j = i & 511;
    G_out[(size_t)(sBase + c) * 512 + j] = Gl[i];
  }
}

// ---------------------------------------------------------------------------
extern "C" void kernel_launch(void* const* d_in, const int* in_sizes, int n_in,
                              void* d_out, int out_size, void* d_ws, size_t ws_size,
                              hipStream_t stream) {
  const float* enc    = (const float*)d_in[0];
  const float* Wh     = (const float*)d_in[1];
  const float* Wu     = (const float*)d_in[2];
  const float* Wpz    = (const float*)d_in[3];
  const float* Wphi1  = (const float*)d_in[4];
  const float* Wphi2  = (const float*)d_in[5];
  const float* Wg     = (const float*)d_in[6];
  const float* gumbel = (const float*)d_in[7];
  const float* unif   = (const float*)d_in[8];

  char* ws = (char*)d_ws;
  bf16_t* encB    = (bf16_t*)(ws + 0);          // 1024x256  bf16
  bf16_t* WhT     = (bf16_t*)(ws + 524288);     // 256x256
  bf16_t* WuT     = (bf16_t*)(ws + 655360);     // 256x256
  bf16_t* W1dT    = (bf16_t*)(ws + 786432);     // 256x256
  bf16_t* WpzTr   = (bf16_t*)(ws + 917504);     // 128x1024
  bf16_t* W1restT = (bf16_t*)(ws + 1179648);    // 256x1152
  bf16_t* WgT     = (bf16_t*)(ws + 1769472);    // 512x256
  bf16_t* usT     = (bf16_t*)(ws + 2031616);    // 256x1024
  bf16_t* hsT     = (bf16_t*)(ws + 2555904);    // 256x1024
  float*  dpart   = (float*)(ws + 3080192);     // 1024x256 f32

  float* out  = (float*)d_out;
  int*   cs   = (int*)out;                      // 128*1024 ints (bit-cast region)
  float* Gout = out + 128 * 1024;               // 128*512 floats

  prep_kernel<<<256, 256, 0, stream>>>(enc, Wh, Wu, Wpz, Wphi1, Wg,
                                       encB, WhT, WuT, W1dT, WpzTr, W1restT, WgT, cs);
  gemm_ew<<<128, 256, 0, stream>>>(encB, WhT, nullptr, hsT, 1);   // hsT (bf16, T)
  gemm_ew<<<128, 256, 0, stream>>>(encB, WuT, nullptr, usT, 1);   // usT (bf16, T)
  gemm_ew<<<128, 256, 0, stream>>>(encB, W1dT, dpart, nullptr, 0);// d_part (f32)
  cluster_kernel<<<8, 512, SMEM_BYTES, stream>>>(enc, Wphi2, gumbel, unif,
                                                 usT, hsT, dpart,
                                                 WpzTr, W1restT, WgT,
                                                 cs, Gout);
}
